// MSDeformAttnPixelDecoder_30442728194522
// MI455X (gfx1250) — compile-verified, matrix operand order checked by reading, not run on hardware
//
#include <hip/hip_runtime.h>
#include <hip/hip_bf16.h>

// ---------------------------------------------------------------------------
// MSDeformAttn pixel decoder for MI455X (gfx1250, wave32).
// All GEMM-shaped work (97% of ~170 GFLOP) routed through
// v_wmma_f32_16x16x32_bf16 (bf16 A/B, fp32 accumulate), 16x64 tile per wave
// (one A fragment feeds 4 WMMAs), branch-free inner loop (all N padded to
// multiples of 64). Irregular work stays fp32 VALU. Transposes are 32x32
// LDS-tiled for coalescing.
// ---------------------------------------------------------------------------

typedef __bf16 bf16;
typedef __attribute__((ext_vector_type(16))) __bf16 v16bf;
typedef __attribute__((ext_vector_type(8)))  float  v8f;

#define LQ     12096
#define CDIM   256
#define AWN    128            // aw projection padded from 96 -> 128 columns

// level geometry: l0=res5(24x24), l1=res4(48x48), l2=res3(96x96)
__device__ __forceinline__ void qlevel(int q, int& l, int& H, int& W, int& base) {
  if (q < 576)       { l = 0; H = 24; W = 24; base = 0;    }
  else if (q < 2880) { l = 1; H = 48; W = 48; base = 576;  }
  else               { l = 2; H = 96; W = 96; base = 2880; }
}

// ISA 16-bit A/B 16x32 fragment: lane = row (mod 16), K pairs striped with the
// half-wave split (k = 16*(v>>2) + 8*(lane>=16) + 2*(v&3)).
__device__ __forceinline__ v16bf load_frag(const bf16* __restrict__ base,
                                           long row, int ld, int k0, int half) {
  v16bf x;
#pragma unroll
  for (int v = 0; v < 8; ++v) {
    const int k = k0 + ((v >> 2) << 4) + (half << 3) + ((v & 3) << 1);
    x[2 * v]     = base[row * ld + k];
    x[2 * v + 1] = base[row * ld + k + 1];
  }
  return x;
}

// ---------------------------------------------------------------------------
// WMMA GEMM: D[M,N] = A[M,K] * W[N,K]^T + bias[N] (optional ReLU).
// Block (32,4): 4 waves stacked along M (64 rows); each wave computes a
// 16x64 strip -> one A fragment feeds 4 WMMAs per k-step, branch-free.
// Requires M%64==0, N%64==0, K%32==0.
// ---------------------------------------------------------------------------
__global__ void __launch_bounds__(128)
gemm_bf16_wmma(const bf16* __restrict__ A, const bf16* __restrict__ W,
               const float* __restrict__ bias, float* __restrict__ D,
               int M, int N, int K, int relu)
{
  const int lane  = threadIdx.x;             // 0..31
  const int wv    = threadIdx.y;             // 0..3
  const int mBase = blockIdx.x * 64 + wv * 16;
  const int nTile = blockIdx.y * 64;
  const int half  = lane >> 4;
  const int l15   = lane & 15;
  const long aRow = (long)(mBase + l15);

  v8f acc[4] = {};
  for (int k0 = 0; k0 < K; k0 += 32) {
    const v16bf a = load_frag(A, aRow, K, k0, half);
#pragma unroll
    for (int j = 0; j < 4; ++j) {
      const v16bf b = load_frag(W, (long)(nTile + 16 * j + l15), K, k0, half);
      acc[j] = __builtin_amdgcn_wmma_f32_16x16x32_bf16(false, a, false, b,
                                                       (short)0, acc[j],
                                                       false, false);
    }
  }
#pragma unroll
  for (int j = 0; j < 4; ++j) {
    const int n  = nTile + 16 * j + l15;
    const float bv = bias ? bias[n] : 0.0f;
#pragma unroll
    for (int i = 0; i < 8; ++i) {
      const int m = mBase + i + (half << 3);   // C/D: m = vgpr + 8*(lane>=16)
      float r = acc[j][i] + bv;
      if (relu) r = fmaxf(r, 0.0f);
      D[(long)m * N + n] = r;
    }
  }
}

// ---------------------------------------------------------------------------
// 3x3 conv as 9 accumulated WMMA GEMMs over zero-padded position-major bf16
// image Ypad[194*194,256]; Wc[9][256][256] tap-major; D[36864,256].
// Same 16x64-per-wave tiling, branch-free.
// ---------------------------------------------------------------------------
__global__ void __launch_bounds__(128)
conv3x3_wmma(const bf16* __restrict__ Ypad, const bf16* __restrict__ Wc,
             float* __restrict__ D)
{
  const int lane  = threadIdx.x;
  const int wv    = threadIdx.y;
  const int mBase = blockIdx.x * 64 + wv * 16;
  const int nTile = blockIdx.y * 64;
  const int half  = lane >> 4;
  const int l15   = lane & 15;
  const int m  = mBase + l15;
  const int my = m / 192, mx = m % 192;

  v8f acc[4] = {};
  for (int t = 0; t < 9; ++t) {
    const int dy = t / 3 - 1, dx = t % 3 - 1;
    const bf16* Arow = Ypad + (long)((my + 1 + dy) * 194 + (mx + 1 + dx)) * 256;
    const bf16* Wt   = Wc + (long)t * 256 * 256;
    for (int k0 = 0; k0 < 256; k0 += 32) {
      const v16bf a = load_frag(Arow, 0, 256, k0, half);
#pragma unroll
      for (int j = 0; j < 4; ++j) {
        const v16bf b = load_frag(Wt, (long)(nTile + 16 * j + l15), 256, k0, half);
        acc[j] = __builtin_amdgcn_wmma_f32_16x16x32_bf16(false, a, false, b,
                                                         (short)0, acc[j],
                                                         false, false);
      }
    }
  }
#pragma unroll
  for (int j = 0; j < 4; ++j) {
    const int n = nTile + 16 * j + l15;
#pragma unroll
    for (int i = 0; i < 8; ++i) {
      const int mm = mBase + i + (half << 3);
      D[(long)mm * 256 + n] = acc[j][i];
    }
  }
}

// ---------------------------------------------------------------------------
// Elementwise / conversion kernels
// ---------------------------------------------------------------------------
__global__ void cvt_f32_bf16(const float* __restrict__ src, bf16* __restrict__ dst, int n) {
  int i = blockIdx.x * 256 + threadIdx.x;
  if (i < n) dst[i] = (bf16)src[i];
}

__global__ void add_cvt_bf16(const float* __restrict__ a, const float* __restrict__ b,
                             bf16* __restrict__ dst, int n) {
  int i = blockIdx.x * 256 + threadIdx.x;
  if (i < n) dst[i] = (bf16)(a[i] + b[i]);
}

// aw_w [6][96][256] -> padded bf16 [6][128][256] (rows 96..127 = 0)
__global__ void cvt_pad_aww(const float* __restrict__ src, bf16* __restrict__ dst) {
  int i = blockIdx.x * 256 + threadIdx.x;
  if (i >= 6 * AWN * 256) return;
  int li = i / (AWN * 256), rem = i % (AWN * 256);
  int n = rem >> 8, k = rem & 255;
  dst[i] = (n < 96) ? (bf16)src[(long)li * 96 * 256 + n * 256 + k] : (bf16)0.0f;
}

// aw_b [6][96] -> padded f32 [6][128]
__global__ void pad_awb(const float* __restrict__ src, float* __restrict__ dst) {
  int i = blockIdx.x * 256 + threadIdx.x;
  if (i >= 6 * AWN) return;
  int li = i / AWN, n = i % AWN;
  dst[i] = (n < 96) ? src[li * 96 + n] : 0.0f;
}

// f [Cin, HW] (channel-major) -> A [HW, Cin] bf16; 32x32 LDS-tiled transpose.
// Grid = (HW/32) * (Cin/32), block 256.
__global__ void __launch_bounds__(256)
transpose_cvt(const float* __restrict__ f, bf16* __restrict__ A, int HW, int Cin) {
  __shared__ float tile[32][33];
  const int tx = threadIdx.x & 31;
  const int ty = threadIdx.x >> 5;          // 0..7
  const int nTilesP = HW >> 5;
  const int tp = (blockIdx.x % nTilesP) << 5;
  const int tk = (blockIdx.x / nTilesP) << 5;
#pragma unroll
  for (int r = 0; r < 32; r += 8)
    tile[ty + r][tx] = f[(long)(tk + ty + r) * HW + tp + tx];     // coalesced p
  __syncthreads();
#pragma unroll
  for (int r = 0; r < 32; r += 8)
    A[(long)(tp + ty + r) * Cin + tk + tx] = (bf16)tile[tx][ty + r]; // coalesced k
}

// X [HW, 256] -> dst [256, HW]; 32x32 LDS-tiled. Grid = (HW/32)*8, block 256.
__global__ void __launch_bounds__(256)
transpose_store(const float* __restrict__ X, float* __restrict__ dst, int HW) {
  __shared__ float tile[32][33];
  const int tx = threadIdx.x & 31;
  const int ty = threadIdx.x >> 5;
  const int nTilesP = HW >> 5;
  const int tp = (blockIdx.x % nTilesP) << 5;
  const int tc = (blockIdx.x / nTilesP) << 5;
#pragma unroll
  for (int r = 0; r < 32; r += 8)
    tile[ty + r][tx] = X[(long)(tp + ty + r) * 256 + tc + tx];    // coalesced c
  __syncthreads();
#pragma unroll
  for (int r = 0; r < 32; r += 8)
    dst[(long)(tc + ty + r) * HW + tp + tx] = tile[tx][ty + r];   // coalesced p
}

// out3_w [O=256][I=256][3][3] -> Wc [9][O][I] bf16
__global__ void permute_conv_w(const float* __restrict__ src, bf16* __restrict__ dst) {
  int i = blockIdx.x * 256 + threadIdx.x;
  if (i >= 9 * 256 * 256) return;
  int t = i >> 16, rem = i & 65535;
  int o = rem >> 8, ii = rem & 255;
  dst[i] = (bf16)src[(long)o * 2304 + ii * 9 + t];
}

// ---------------------------------------------------------------------------
// Sine positional embedding + level embedding -> pos[Lq, 256]
// ---------------------------------------------------------------------------
__global__ void pos_embed_kernel(float* __restrict__ pos, const float* __restrict__ lvl_emb) {
  int i = blockIdx.x * 256 + threadIdx.x;
  if (i >= LQ * CDIM) return;
  int q = i >> 8, c = i & 255;
  int l, H, W, base; qlevel(q, l, H, W, base);
  int local = q - base;
  int yy = local / W, xx = local - yy * W;
  const float twopi = 6.283185307179586f;
  float e, j;
  if (c < 128) { e = (yy + 1) / ((float)H + 1e-6f) * twopi; j = (float)c; }
  else         { e = (xx + 1) / ((float)W + 1e-6f) * twopi; j = (float)(c - 128); }
  float dim_t = __powf(10000.0f, 2.0f * floorf(j * 0.5f) / 128.0f);
  float p = e / dim_t;
  float v = (((int)j) & 1) ? __cosf(p) : __sinf(p);
  pos[i] = v + lvl_emb[l * CDIM + c];
}

// ---------------------------------------------------------------------------
// GroupNorm over position-major X[HW,256], 32 groups of 8 channels.
// grid = 32 blocks, 256 threads. Optional ReLU.
// ---------------------------------------------------------------------------
__global__ void __launch_bounds__(256)
group_norm_kernel(float* __restrict__ X, int HW,
                  const float* __restrict__ gamma, const float* __restrict__ beta,
                  int relu)
{
  const int grp = blockIdx.x;      // 0..31
  const int tid = threadIdx.x;
  __shared__ float s1[256], s2[256];
  __shared__ float mu, rs;
  float sum = 0.f, sq = 0.f;
  for (int p = tid; p < HW; p += 256) {
    const float* r = X + (long)p * 256 + grp * 8;
#pragma unroll
    for (int c = 0; c < 8; ++c) { float v = r[c]; sum += v; sq += v * v; }
  }
  s1[tid] = sum; s2[tid] = sq; __syncthreads();
  for (int s = 128; s > 0; s >>= 1) {
    if (tid < s) { s1[tid] += s1[tid + s]; s2[tid] += s2[tid + s]; }
    __syncthreads();
  }
  if (tid == 0) {
    float n = (float)HW * 8.0f;
    float m = s1[0] / n;
    float var = s2[0] / n - m * m;
    mu = m; rs = rsqrtf(var + 1e-5f);
  }
  __syncthreads();
  for (int p = tid; p < HW; p += 256) {
    float* r = X + (long)p * 256 + grp * 8;
#pragma unroll
    for (int c = 0; c < 8; ++c) {
      int ch = grp * 8 + c;
      float v = (r[c] - mu) * rs * gamma[ch] + beta[ch];
      if (relu) v = fmaxf(v, 0.0f);
      r[c] = v;
    }
  }
}

// ---------------------------------------------------------------------------
// src = LayerNorm(src + delta) over 256-wide rows. One wave32 per row.
// ---------------------------------------------------------------------------
__global__ void __launch_bounds__(32)
add_layernorm_kernel(float* __restrict__ src, const float* __restrict__ delta,
                     const float* __restrict__ g, const float* __restrict__ b)
{
  const int row = blockIdx.x, lane = threadIdx.x;
  const long base = (long)row * 256;
  float v[8]; float sum = 0.f, sq = 0.f;
#pragma unroll
  for (int j = 0; j < 8; ++j) {
    int c = lane * 8 + j;
    float x = src[base + c] + delta[base + c];
    v[j] = x; sum += x; sq += x * x;
  }
  for (int off = 16; off > 0; off >>= 1) {
    sum += __shfl_xor(sum, off, 32);
    sq  += __shfl_xor(sq,  off, 32);
  }
  float mu = sum * (1.0f / 256.0f);
  float var = sq * (1.0f / 256.0f) - mu * mu;
  float rs = rsqrtf(var + 1e-5f);
#pragma unroll
  for (int j = 0; j < 8; ++j) {
    int c = lane * 8 + j;
    src[base + c] = (v[j] - mu) * rs * g[c] + b[c];
  }
}

// softmax over 12 values (NLVL*NPTS) per (q,h); aw row stride = AWN(128)
__global__ void softmax12_kernel(float* __restrict__ aw, int total) {
  int i = blockIdx.x * 256 + threadIdx.x;
  if (i >= total) return;
  int q = i >> 3, h = i & 7;
  float* p = aw + (long)q * AWN + h * 12;
  float m = -1e30f;
#pragma unroll
  for (int j = 0; j < 12; ++j) m = fmaxf(m, p[j]);
  float s = 0.f;
#pragma unroll
  for (int j = 0; j < 12; ++j) { float e = __expf(p[j] - m); p[j] = e; s += e; }
  float inv = 1.0f / s;
#pragma unroll
  for (int j = 0; j < 12; ++j) p[j] *= inv;
}

// ---------------------------------------------------------------------------
// Deformable attention sampling: attn[q,c] = sum_{l,p} aw * bilinear(v)
// One thread per (q, c); 12 bilinear gathers each (zero padding).
// ---------------------------------------------------------------------------
__global__ void __launch_bounds__(256)
deform_attn_kernel(const float* __restrict__ v, const float* __restrict__ off,
                   const float* __restrict__ aw, float* __restrict__ attn)
{
  int i = blockIdx.x * 256 + threadIdx.x;
  if (i >= LQ * CDIM) return;
  int q = i >> 8, c = i & 255;
  int h = c >> 5;
  int lq, H, W, base; qlevel(q, lq, H, W, base);
  int local = q - base;
  float ry = ((local / W) + 0.5f) / (float)H;
  float rx = ((local % W) + 0.5f) / (float)W;

  const int LS[3] = {24, 48, 96};
  const int LB[3] = {0, 576, 2880};
  float acc = 0.0f;
#pragma unroll
  for (int l = 0; l < 3; ++l) {
    const int S = LS[l], st = LB[l];
    const float Sf = (float)S;
#pragma unroll
    for (int p = 0; p < 4; ++p) {
      const long oi = ((((long)q * 8 + h) * 3 + l) * 4 + p) * 2;
      float lx = rx + off[oi]     / Sf;   // offset_norm = (W_l, H_l)
      float ly = ry + off[oi + 1] / Sf;
      float x = lx * Sf - 0.5f, y = ly * Sf - 0.5f;
      float xf = floorf(x), yf = floorf(y);
      int ix = (int)xf, iy = (int)yf;
      float wx = x - xf, wy = y - yf;
      float s00 = 0.f, s10 = 0.f, s01 = 0.f, s11 = 0.f;
      bool vx0 = (ix >= 0) & (ix < S),     vx1 = (ix + 1 >= 0) & (ix + 1 < S);
      bool vy0 = (iy >= 0) & (iy < S),     vy1 = (iy + 1 >= 0) & (iy + 1 < S);
      if (vx0 & vy0) s00 = v[(long)(st + iy * S + ix) * 256 + c];
      if (vx1 & vy0) s10 = v[(long)(st + iy * S + ix + 1) * 256 + c];
      if (vx0 & vy1) s01 = v[(long)(st + (iy + 1) * S + ix) * 256 + c];
      if (vx1 & vy1) s11 = v[(long)(st + (iy + 1) * S + ix + 1) * 256 + c];
      float bil = s00 * (1 - wx) * (1 - wy) + s10 * wx * (1 - wy)
                + s01 * (1 - wx) * wy       + s11 * wx * wy;
      acc += aw[(long)q * AWN + h * 12 + l * 4 + p] * bil;
    }
  }
  attn[i] = acc;
}

// half-pixel bilinear upsample of src level-2 (96x96) to 192x192, add to cur
__global__ void upsample_add_kernel(float* __restrict__ cur, const float* __restrict__ src) {
  int i = blockIdx.x * 256 + threadIdx.x;
  if (i >= 36864 * 256) return;
  int p = i >> 8, c = i & 255;
  int Y = p / 192, X = p - Y * 192;
  float sy = (Y + 0.5f) * 0.5f - 0.5f;
  float sx = (X + 0.5f) * 0.5f - 0.5f;
  float yf = floorf(sy), xf = floorf(sx);
  int y0 = (int)yf, x0 = (int)xf;
  float wy = sy - yf, wx = sx - xf;
  int y0c = min(max(y0, 0), 95), y1c = min(max(y0 + 1, 0), 95);
  int x0c = min(max(x0, 0), 95), x1c = min(max(x0 + 1, 0), 95);
  const float* S = src + (long)2880 * 256;
  float v00 = S[(long)(y0c * 96 + x0c) * 256 + c];
  float v10 = S[(long)(y0c * 96 + x1c) * 256 + c];
  float v01 = S[(long)(y1c * 96 + x0c) * 256 + c];
  float v11 = S[(long)(y1c * 96 + x1c) * 256 + c];
  cur[i] += v00 * (1 - wx) * (1 - wy) + v10 * wx * (1 - wy)
          + v01 * (1 - wx) * wy       + v11 * wx * wy;
}

// zero-pad [192,192,256] -> [194,194,256] bf16
__global__ void pad_cvt_kernel(const float* __restrict__ y0, bf16* __restrict__ ypad) {
  int i = blockIdx.x * 256 + threadIdx.x;
  if (i >= 194 * 194 * 256) return;
  int r = i >> 8, c = i & 255;
  int ry = r / 194, rx = r - ry * 194;
  bf16 v = (bf16)0.0f;
  if (ry >= 1 && ry <= 192 && rx >= 1 && rx <= 192)
    v = (bf16)y0[(long)((ry - 1) * 192 + (rx - 1)) * 256 + c];
  ypad[i] = v;
}

// ---------------------------------------------------------------------------
// Host side
// ---------------------------------------------------------------------------
static inline dim3 gemm_grid(int M, int N) { return dim3(M / 64, N / 64); }

extern "C" void kernel_launch(void* const* d_in, const int* in_sizes, int n_in,
                              void* d_out, int out_size, void* d_ws, size_t ws_size,
                              hipStream_t stream) {
  (void)in_sizes; (void)n_in; (void)out_size; (void)ws_size;
  // ---- inputs (setup_inputs order) ----
  const float* f_res2   = (const float*)d_in[0];
  const float* f_res3   = (const float*)d_in[1];
  const float* f_res4   = (const float*)d_in[2];
  const float* f_res5   = (const float*)d_in[3];
  const float* pw5      = (const float*)d_in[4];
  const float* pw4      = (const float*)d_in[5];
  const float* pw3      = (const float*)d_in[6];
  const float* proj_b   = (const float*)d_in[7];
  const float* proj_gg  = (const float*)d_in[8];
  const float* proj_gb  = (const float*)d_in[9];
  const float* lvl_emb  = (const float*)d_in[10];
  const float* off_w    = (const float*)d_in[11];
  const float* off_b    = (const float*)d_in[12];
  const float* aw_w     = (const float*)d_in[13];
  const float* aw_b     = (const float*)d_in[14];
  const float* vp_w     = (const float*)d_in[15];
  const float* vp_b     = (const float*)d_in[16];
  const float* op_w     = (const float*)d_in[17];
  const float* op_b     = (const float*)d_in[18];
  const float* ln1_g    = (const float*)d_in[19];
  const float* ln1_b    = (const float*)d_in[20];
  const float* ff1_w    = (const float*)d_in[21];
  const float* ff1_b    = (const float*)d_in[22];
  const float* ff2_w    = (const float*)d_in[23];
  const float* ff2_b    = (const float*)d_in[24];
  const float* ln2_g    = (const float*)d_in[25];
  const float* ln2_b    = (const float*)d_in[26];
  const float* lat_w    = (const float*)d_in[27];
  const float* lat_gg   = (const float*)d_in[28];
  const float* lat_gb   = (const float*)d_in[29];
  const float* out3_w   = (const float*)d_in[30];
  const float* out_gg   = (const float*)d_in[31];
  const float* out_gb   = (const float*)d_in[32];
  const float* mask_w   = (const float*)d_in[33];
  const float* mask_b   = (const float*)d_in[34];

  // ---- workspace carving ----
  size_t cur_off = 0;
  auto alloc = [&](size_t bytes) -> void* {
    size_t o = (cur_off + 255) & ~(size_t)255;
    cur_off = o + bytes;
    return (void*)((char*)d_ws + o);
  };
  const int HW2 = 36864;                       // 192*192
  float* srcf   = (float*)alloc((size_t)LQ * 256 * 4);
  float* posf   = (float*)alloc((size_t)LQ * 256 * 4);
  float* offf   = (float*)alloc((size_t)LQ * 384 * 4);
  float* awf    = (float*)alloc((size_t)LQ * AWN * 4);
  float* vf     = (float*)alloc((size_t)LQ * 256 * 4);
  float* attnf  = (float*)alloc((size_t)LQ * 256 * 4);
  float* tmpf   = (float*)alloc((size_t)LQ * 256 * 4);
  float* hf     = (float*)alloc((size_t)LQ * 1024 * 4);
  bf16*  qb     = (bf16*)alloc((size_t)LQ * 256 * 2);
  bf16*  srcb   = (bf16*)alloc((size_t)LQ * 256 * 2);
  bf16*  hb     = (bf16*)alloc((size_t)LQ * 1024 * 2);
  bf16*  Abuf   = (bf16*)alloc((size_t)HW2 * 256 * 2);   // max transposed input
  bf16*  pw5b   = (bf16*)alloc((size_t)256 * 2048 * 2);
  bf16*  pw4b   = (bf16*)alloc((size_t)256 * 1024 * 2);
  bf16*  pw3b   = (bf16*)alloc((size_t)256 * 512 * 2);
  bf16*  offwb  = (bf16*)alloc((size_t)6 * 384 * 256 * 2);
  bf16*  awwb   = (bf16*)alloc((size_t)6 * AWN * 256 * 2);   // zero-padded
  float* awbp   = (float*)alloc((size_t)6 * AWN * 4);        // zero-padded bias
  bf16*  vpwb   = (bf16*)alloc((size_t)6 * 256 * 256 * 2);
  bf16*  opwb   = (bf16*)alloc((size_t)6 * 256 * 256 * 2);
  bf16*  ff1wb  = (bf16*)alloc((size_t)6 * 1024 * 256 * 2);
  bf16*  ff2wb  = (bf16*)alloc((size_t)6 * 256 * 1024 * 2);
  bf16*  latwb  = (bf16*)alloc((size_t)256 * 256 * 2);
  bf16*  maskwb = (bf16*)alloc((size_t)256 * 256 * 2);
  bf16*  convwb = (bf16*)alloc((size_t)9 * 256 * 256 * 2);
  float* curf   = (float*)alloc((size_t)HW2 * 256 * 4);
  bf16*  ypad   = (bf16*)alloc((size_t)194 * 194 * 256 * 2);
  float* convf  = (float*)alloc((size_t)HW2 * 256 * 4);
  bf16*  ybf    = (bf16*)alloc((size_t)HW2 * 256 * 2);
  float* maskf  = (float*)alloc((size_t)HW2 * 256 * 4);

  auto CVT = [&](const float* s, bf16* d, int n) {
    cvt_f32_bf16<<<(n + 255) / 256, 256, 0, stream>>>(s, d, n);
  };

  // ---- weight conversion to bf16 ----
  CVT(pw5, pw5b, 256 * 2048);
  CVT(pw4, pw4b, 256 * 1024);
  CVT(pw3, pw3b, 256 * 512);
  CVT(off_w, offwb, 6 * 384 * 256);
  cvt_pad_aww<<<(6 * AWN * 256 + 255) / 256, 256, 0, stream>>>(aw_w, awwb);
  pad_awb<<<(6 * AWN + 255) / 256, 256, 0, stream>>>(aw_b, awbp);
  CVT(vp_w,  vpwb,  6 * 256 * 256);
  CVT(op_w,  opwb,  6 * 256 * 256);
  CVT(ff1_w, ff1wb, 6 * 1024 * 256);
  CVT(ff2_w, ff2wb, 6 * 256 * 1024);
  CVT(lat_w, latwb, 256 * 256);
  CVT(mask_w, maskwb, 256 * 256);
  permute_conv_w<<<(9 * 256 * 256 + 255) / 256, 256, 0, stream>>>(out3_w, convwb);

  // ---- input projections + GroupNorm (level order: res5, res4, res3) ----
  const int   HWs[3]  = {576, 2304, 9216};
  const int   Cins[3] = {2048, 1024, 512};
  const int   starts[3] = {0, 576, 2880};
  const float* fs[3]  = {f_res5, f_res4, f_res3};
  const bf16*  pws[3] = {pw5b, pw4b, pw3b};
  for (int l = 0; l < 3; ++l) {
    int HW = HWs[l], Cin = Cins[l];
    transpose_cvt<<<(HW / 32) * (Cin / 32), 256, 0, stream>>>(fs[l], Abuf, HW, Cin);
    gemm_bf16_wmma<<<gemm_grid(HW, 256), dim3(32, 4), 0, stream>>>(
        Abuf, pws[l], proj_b + l * 256, srcf + (long)starts[l] * 256, HW, 256, Cin, 0);
    group_norm_kernel<<<32, 256, 0, stream>>>(srcf + (long)starts[l] * 256, HW,
                                              proj_gg + l * 256, proj_gb + l * 256, 0);
  }

  // ---- positional embeddings ----
  pos_embed_kernel<<<(LQ * 256 + 255) / 256, 256, 0, stream>>>(posf, lvl_emb);

  // ---- 6 deformable encoder layers ----
  for (int li = 0; li < 6; ++li) {
    add_cvt_bf16<<<(LQ * 256 + 255) / 256, 256, 0, stream>>>(srcf, posf, qb, LQ * 256);
    gemm_bf16_wmma<<<gemm_grid(LQ, 384), dim3(32, 4), 0, stream>>>(
        qb, offwb + (long)li * 384 * 256, off_b + li * 384, offf, LQ, 384, 256, 0);
    gemm_bf16_wmma<<<gemm_grid(LQ, AWN), dim3(32, 4), 0, stream>>>(
        qb, awwb + (long)li * AWN * 256, awbp + li * AWN, awf, LQ, AWN, 256, 0);
    softmax12_kernel<<<(LQ * 8 + 255) / 256, 256, 0, stream>>>(awf, LQ * 8);
    CVT(srcf, srcb, LQ * 256);
    gemm_bf16_wmma<<<gemm_grid(LQ, 256), dim3(32, 4), 0, stream>>>(
        srcb, vpwb + (long)li * 256 * 256, vp_b + li * 256, vf, LQ, 256, 256, 0);
    deform_attn_kernel<<<(LQ * 256 + 255) / 256, 256, 0, stream>>>(vf, offf, awf, attnf);
    CVT(attnf, srcb, LQ * 256);   // reuse srcb as bf16 attn
    gemm_bf16_wmma<<<gemm_grid(LQ, 256), dim3(32, 4), 0, stream>>>(
        srcb, opwb + (long)li * 256 * 256, op_b + li * 256, tmpf, LQ, 256, 256, 0);
    add_layernorm_kernel<<<LQ, 32, 0, stream>>>(srcf, tmpf, ln1_g + li * 256, ln1_b + li * 256);
    CVT(srcf, srcb, LQ * 256);
    gemm_bf16_wmma<<<gemm_grid(LQ, 1024), dim3(32, 4), 0, stream>>>(
        srcb, ff1wb + (long)li * 1024 * 256, ff1_b + li * 1024, hf, LQ, 1024, 256, 1);
    CVT(hf, hb, LQ * 1024);
    gemm_bf16_wmma<<<gemm_grid(LQ, 256), dim3(32, 4), 0, stream>>>(
        hb, ff2wb + (long)li * 256 * 1024, ff2_b + li * 256, tmpf, LQ, 256, 1024, 0);
    add_layernorm_kernel<<<LQ, 32, 0, stream>>>(srcf, tmpf, ln2_g + li * 256, ln2_b + li * 256);
  }

  // ---- FPN tail ----
  transpose_cvt<<<(HW2 / 32) * (256 / 32), 256, 0, stream>>>(f_res2, Abuf, HW2, 256);
  gemm_bf16_wmma<<<gemm_grid(HW2, 256), dim3(32, 4), 0, stream>>>(
      Abuf, latwb, nullptr, curf, HW2, 256, 256, 0);
  group_norm_kernel<<<32, 256, 0, stream>>>(curf, HW2, lat_gg, lat_gb, 0);
  upsample_add_kernel<<<(HW2 * 256 + 255) / 256, 256, 0, stream>>>(curf, srcf);
  pad_cvt_kernel<<<(194 * 194 * 256 + 255) / 256, 256, 0, stream>>>(curf, ypad);
  conv3x3_wmma<<<dim3(HW2 / 64, 4), dim3(32, 4), 0, stream>>>(ypad, convwb, convf);
  group_norm_kernel<<<32, 256, 0, stream>>>(convf, HW2, out_gg, out_gb, 1);
  CVT(convf, ybf, HW2 * 256);
  gemm_bf16_wmma<<<gemm_grid(HW2, 256), dim3(32, 4), 0, stream>>>(
      ybf, maskwb, mask_b, maskf, HW2, 256, 256, 0);

  // ---- outputs: (mask, out0, (out0, out1, out2)) flattened ----
  float* out = (float*)d_out;
  long o = 0;
  transpose_store<<<(HW2 / 32) * 8, 256, 0, stream>>>(maskf, out + o, HW2);
  o += (long)HW2 * 256;
  transpose_store<<<(576 / 32) * 8, 256, 0, stream>>>(srcf, out + o, 576);
  o += 576 * 256;
  transpose_store<<<(576 / 32) * 8, 256, 0, stream>>>(srcf, out + o, 576);
  o += 576 * 256;
  transpose_store<<<(2304 / 32) * 8, 256, 0, stream>>>(
      srcf + (long)576 * 256, out + o, 2304);
  o += 2304 * 256;
  transpose_store<<<(9216 / 32) * 8, 256, 0, stream>>>(
      srcf + (long)2880 * 256, out + o, 9216);
}
